// AttentionNet_79972291051647
// MI455X (gfx1250) — compile-verified, matrix operand order checked by reading, not run on hardware
//
#include <hip/hip_runtime.h>
#include <math.h>

// ---------------------------------------------------------------------------
// AttentionNet for MI455X (gfx1250, wave32, WMMA bf16 16x16x32, f32 accum)
// N = B*T = 2560 items; per item: vis[49,512] @ Wv^T[512,512] (dominant GEMM),
// v_t[49,512] @ Av^T[512,49], plus small VALU phases.
// LDS tiles use a padded row stride (520 bf16 = 260 dwords, 260 % 64 == 4) so
// WMMA fragment ds_load_b128 gathers are bank-conflict-free.
// ---------------------------------------------------------------------------

typedef __attribute__((ext_vector_type(16))) __bf16 bf16x16;
typedef __attribute__((ext_vector_type(8)))  float  f32x8;
typedef __attribute__((ext_vector_type(4)))  unsigned int u32x4;

#define NITEMS 2560
#define SDIM   49
#define SPAD   64
#define DVIS   512
#define DAUD   128
#define EDIM   512
#define LSTR   520     // padded LDS row stride (bf16 elems), 16B-aligned rows
#define VS_STRIDE 65

__device__ __forceinline__ unsigned short f2bf(float f) {
  union { float f; unsigned u; } v; v.f = f;
  unsigned b = v.u;
  return (unsigned short)((b + 0x7FFFu + ((b >> 16) & 1u)) >> 16);
}

// Load a 16-element bf16 fragment for A/B operands of wmma_f32_16x16x32_bf16.
// Per ISA layout: lane group g = lane>>4 selects k-block; halves 0..7 are at
// element offset +0..7, halves 8..15 at +16..23 -> two 16B loads.
__device__ __forceinline__ bf16x16 load_frag(const unsigned short* p) {
  union { u32x4 u[2]; bf16x16 v; } cv;
  cv.u[0] = *(const u32x4*)(p);
  cv.u[1] = *(const u32x4*)(p + 16);
  return cv.v;
}

// ---------------------------------------------------------------------------
// Prep: Wv f32 -> bf16, Av f32 -> bf16 zero-padded to [64,512]
// ---------------------------------------------------------------------------
__global__ __launch_bounds__(256) void prep_kernel(
    const float* __restrict__ Wv, const float* __restrict__ Av,
    unsigned short* __restrict__ wvb, unsigned short* __restrict__ avb) {
  int idx = blockIdx.x * blockDim.x + threadIdx.x;
  if (idx < EDIM * DVIS) {
    wvb[idx] = f2bf(Wv[idx]);
  }
  int j = idx - EDIM * DVIS;
  if (j >= 0 && j < SPAD * EDIM) {
    int f = j >> 9, e = j & 511;
    avb[j] = (f < SDIM) ? f2bf(Av[f * EDIM + e]) : (unsigned short)0;
  }
}

// ---------------------------------------------------------------------------
// Main: one workgroup (256 threads = 8 wave32) per item n
// ---------------------------------------------------------------------------
__global__ __launch_bounds__(256) void attn_kernel(
    const float* __restrict__ audio, const float* __restrict__ visual,
    const float* __restrict__ Wa, const float* __restrict__ ba,
    const float* __restrict__ bv,
    const float* __restrict__ Aa, const float* __restrict__ Af,
    const unsigned short* __restrict__ wvb,
    const unsigned short* __restrict__ avb,
    float* __restrict__ out) {
  extern __shared__ unsigned char smem_raw[];
  unsigned short* visb = (unsigned short*)smem_raw;          // [64][520] bf16
  unsigned short* vtb  = visb + SPAD * LSTR;                 // [64][520] bf16
  float* vs  = (float*)(vtb + SPAD * LSTR);                  // [64][65]  f32
  float* aud = vs + SPAD * VS_STRIDE;                        // [128]
  float* a_t = aud + DAUD;                                   // [512]
  float* a_s = a_t + EDIM;                                   // [64]
  float* fv  = a_s + SPAD;                                   // [64]
  float* att = fv + SPAD;                                    // [64]
  float* red = att + SPAD;                                   // [4]

  const int n    = blockIdx.x;
  const int t    = threadIdx.x;
  const int lane = t & 31;
  const int wave = t >> 5;
  const int r    = lane & 15;
  const int g    = lane >> 4;

  const float* visn = visual + (size_t)n * SDIM * DVIS;

  // ---- Phase 0: stage vis (bf16, zero-padded rows) + audio into LDS --------
  for (int idx = t; idx < SPAD * DVIS; idx += 256) {
    int s = idx >> 9, d = idx & 511;
    visb[s * LSTR + d] = (s < SDIM) ? f2bf(visn[s * DVIS + d]) : (unsigned short)0;
  }
  if (t < DAUD) aud[t] = audio[(size_t)n * DAUD + t];
  __syncthreads();

  // ---- Phase 1: a_t = relu(aud @ Wa^T + ba), 512 outputs, 2 per thread -----
  for (int e = t; e < EDIM; e += 256) {
    float acc = ba[e];
    const float* w = Wa + (size_t)e * DAUD;
#pragma unroll 4
    for (int d = 0; d < DAUD; ++d) acc = fmaf(aud[d], w[d], acc);
    a_t[e] = fmaxf(acc, 0.f);
  }

  // ---- Phase 2: v_t = relu(vis @ Wv^T + bv) via WMMA -----------------------
  // M=64 (s, padded), N=512 (e), K=512 (d). Each wave: 4 N-tiles.
  for (int nt = wave * 4; nt < wave * 4 + 4; ++nt) {
    const int ncol = nt * 16;
    // Prefetch the next N-tile's B rows toward this WGP while we compute.
    if (nt + 1 < wave * 4 + 4)
      __builtin_prefetch(wvb + (size_t)(ncol + 16 + r) * DVIS + g * 256, 0, 0);

    const float bvv = bv[ncol + r];
    f32x8 acc0 = {bvv, bvv, bvv, bvv, bvv, bvv, bvv, bvv};
    f32x8 acc1 = acc0, acc2 = acc0, acc3 = acc0;
    for (int kt = 0; kt < 16; ++kt) {
      const int kb = kt * 32 + g * 8;
      bf16x16 bfr = load_frag(wvb + (size_t)(ncol + r) * DVIS + kb);  // B: Wv[e][d]
      bf16x16 a0 = load_frag(visb + (0 * 16 + r) * LSTR + kb);
      bf16x16 a1 = load_frag(visb + (1 * 16 + r) * LSTR + kb);
      bf16x16 a2 = load_frag(visb + (2 * 16 + r) * LSTR + kb);
      bf16x16 a3 = load_frag(visb + (3 * 16 + r) * LSTR + kb);
      acc0 = __builtin_amdgcn_wmma_f32_16x16x32_bf16(false, a0, false, bfr, (short)0, acc0, false, false);
      acc1 = __builtin_amdgcn_wmma_f32_16x16x32_bf16(false, a1, false, bfr, (short)0, acc1, false, false);
      acc2 = __builtin_amdgcn_wmma_f32_16x16x32_bf16(false, a2, false, bfr, (short)0, acc2, false, false);
      acc3 = __builtin_amdgcn_wmma_f32_16x16x32_bf16(false, a3, false, bfr, (short)0, acc3, false, false);
    }
#pragma unroll
    for (int i = 0; i < 8; ++i) {
      const int ro = i + 8 * g;  // C layout: VGPR i -> row i (+8 for lanes 16-31)
      vtb[(0 * 16 + ro) * LSTR + ncol + r] = f2bf(fmaxf(acc0[i], 0.f));
      vtb[(1 * 16 + ro) * LSTR + ncol + r] = f2bf(fmaxf(acc1[i], 0.f));
      vtb[(2 * 16 + ro) * LSTR + ncol + r] = f2bf(fmaxf(acc2[i], 0.f));
      vtb[(3 * 16 + ro) * LSTR + ncol + r] = f2bf(fmaxf(acc3[i], 0.f));
    }
  }
  __syncthreads();

  // ---- Phase 3: a_s[s] = a_t . Aa[s,:]  (49 rows) --------------------------
  if (t < SDIM) {
    float acc = 0.f;
    const float* aa = Aa + (size_t)t * EDIM;
#pragma unroll 4
    for (int e = 0; e < EDIM; ++e) acc = fmaf(a_t[e], aa[e], acc);
    a_s[t] = acc;
  }
  __syncthreads();

  // ---- Phase 4: v_s = v_t @ Av^T via WMMA (M=64,N=64,K=512) ----------------
  for (int p = wave * 2; p < wave * 2 + 2; ++p) {
    const int mt = p >> 2, nt = p & 3;
    f32x8 acc = {0.f, 0.f, 0.f, 0.f, 0.f, 0.f, 0.f, 0.f};
    for (int kt = 0; kt < 16; ++kt) {
      const int kb = kt * 32 + g * 8;
      bf16x16 a = load_frag(vtb + (mt * 16 + r) * LSTR + kb);
      bf16x16 b = load_frag(avb + (size_t)(nt * 16 + r) * EDIM + kb);  // B: Av[f][e]
      acc = __builtin_amdgcn_wmma_f32_16x16x32_bf16(false, a, false, b, (short)0, acc, false, false);
    }
#pragma unroll
    for (int i = 0; i < 8; ++i)
      vs[(mt * 16 + i + 8 * g) * VS_STRIDE + nt * 16 + r] = acc[i];
  }
  __syncthreads();

  // ---- Phase 5: f[s] = sum_f tanh(a_s[s] + v_s[s,f]) * Af[f] ---------------
  if (t < SDIM) {
    const float m = a_s[t];
    float acc = 0.f;
    for (int f = 0; f < SDIM; ++f)
      acc = fmaf(tanhf(m + vs[t * VS_STRIDE + f]), Af[f], acc);
    fv[t] = acc;
  }
  __syncthreads();

  // ---- Phase 6: softmax over s ---------------------------------------------
  if (t == 0) {
    float mx = fv[0];
    for (int s = 1; s < SDIM; ++s) mx = fmaxf(mx, fv[s]);
    red[0] = mx;
  }
  __syncthreads();
  if (t < SDIM) att[t] = expf(fv[t] - red[0]);
  __syncthreads();
  if (t == 0) {
    float sm = 0.f;
    for (int s = 0; s < SDIM; ++s) sm += att[s];
    red[1] = 1.f / sm;
  }
  __syncthreads();

  // ---- Phase 7: out[d] = sum_s att[s] * vis[s,d]  (f32 from global/L2) -----
  const float inv = red[1];
#pragma unroll
  for (int dd = 0; dd < 2; ++dd) {
    const int d = t + dd * 256;
    float acc = 0.f;
    for (int s = 0; s < SDIM; ++s)
      acc = fmaf(att[s] * inv, visn[s * DVIS + d], acc);
    out[(size_t)n * DVIS + d] = acc;
  }
}

// ---------------------------------------------------------------------------
extern "C" void kernel_launch(void* const* d_in, const int* in_sizes, int n_in,
                              void* d_out, int out_size, void* d_ws, size_t ws_size,
                              hipStream_t stream) {
  const float* audio  = (const float*)d_in[0];
  const float* visual = (const float*)d_in[1];
  const float* Wa     = (const float*)d_in[2];
  const float* ba     = (const float*)d_in[3];
  const float* Wv     = (const float*)d_in[4];
  const float* bv     = (const float*)d_in[5];
  const float* Aa     = (const float*)d_in[6];
  const float* Av     = (const float*)d_in[7];
  const float* Af     = (const float*)d_in[8];
  float* out = (float*)d_out;

  unsigned short* wvb = (unsigned short*)d_ws;        // 512*512 bf16
  unsigned short* avb = wvb + EDIM * DVIS;            // 64*512 bf16 (padded)

  const int prep_n = EDIM * DVIS + SPAD * EDIM;
  prep_kernel<<<(prep_n + 255) / 256, 256, 0, stream>>>(Wv, Av, wvb, avb);

  const size_t smem =
      (size_t)SPAD * LSTR * 2 +           // visb bf16 (padded stride)
      (size_t)SPAD * LSTR * 2 +           // vtb  bf16 (padded stride)
      (size_t)SPAD * VS_STRIDE * 4 +      // vs   f32
      (DAUD + EDIM + SPAD * 3 + 4) * 4;   // aud, a_t, a_s, fv, att, red

  attn_kernel<<<NITEMS, 256, smem, stream>>>(audio, visual, Wa, ba, bv, Aa, Af,
                                             wvb, avb, out);
}
